// DualOutputRNN_283467842338
// MI455X (gfx1250) — compile-verified, matrix operand order checked by reading, not run on hardware
//
#include <hip/hip_runtime.h>
#include <hip/hip_bf16.h>

typedef __attribute__((ext_vector_type(16))) _Float16 v16h;
typedef __attribute__((ext_vector_type(8)))  float    v8f;

#define B_     4096
#define T_     256
#define CIN_   16
#define HID_   64
#define NCLS_  10
#define KW_    80     // C_IN + HID
#define NT_    16     // 256 gate outputs / 16
#define KT_    3      // K padded to 96 = 3 tiles of 32

#define LOG2E_ 1.4426950408889634f
#define LN2_   0.6931471805599453f

// WMMA 16-bit fragment index helpers (ISA 7.12.2 layout):
// within a 32-wide K tile: element e of lane-half h holds k = ((e>>3)<<4) | (h<<3) | (e&7)
__device__ __forceinline__ int eOfK(int kr){ return ((kr >> 4) << 3) | (kr & 7); }
__device__ __forceinline__ int hOfK(int kr){ return (kr >> 3) & 1; }

// Branch-free transcendentals on v_exp_f32 / v_rcp_f32 / v_log_f32
__device__ __forceinline__ float fast_sig(float x) {
    float t = __builtin_amdgcn_exp2f(-x * LOG2E_);      // exp(-x); inf for very negative x
    return __builtin_amdgcn_rcpf(1.f + t);              // rcp(inf) -> 0, correct limit
}
__device__ __forceinline__ float fast_tanh(float x) {
    float t = __builtin_amdgcn_exp2f(-2.f * LOG2E_ * __builtin_fabsf(x)); // (0,1]
    float r = (1.f - t) * __builtin_amdgcn_rcpf(1.f + t);
    return __builtin_copysignf(r, x);
}

__global__ __launch_bounds__(128)
void convlstm_fused(const float* __restrict__ x,
                    const float* __restrict__ Wl,
                    const float* __restrict__ bl,
                    const float* __restrict__ Wc,
                    const float* __restrict__ bc,
                    const float* __restrict__ Wd,
                    const float* __restrict__ bd,
                    float* __restrict__ out_logp,
                    float* __restrict__ out_pt)
{
    __shared__ _Float16 wfrag[NT_ * KT_ * 512]; // W_lstm^T in B-fragment layout, 48 KB
    __shared__ _Float16 comb [KT_ * 512];       // [x_t | h] A-fragment layout (K=96), 3 KB
    __shared__ _Float16 hbuf [2 * 512];         // h in classifier A layout (K=64),   2 KB

    const int tid = threadIdx.x;
    const int l   = tid & 31;          // lane
    const int hw  = tid >> 5;          // wave id = hidden slice 0..3
    const int n   = l & 15;            // output column within tile
    const int mhi = (l >> 4) * 8;      // +8 rows for upper half-lanes (C/D layout)
    const int b0  = blockIdx.x * 16;   // batch tile base

    // ---- one-time: convert W_lstm^T into f16 B fragments in LDS ----
    for (int idx = tid; idx < NT_ * KT_ * 512; idx += 128) {
        int e  = idx & 15;
        int ll = (idx >> 4) & 31;
        int g  = idx >> 9;             // nt*3 + kk
        int kk = g % 3, nt = g / 3;
        int nn = nt * 16 + (ll & 15);
        int k  = kk * 32 + (((e >> 3) << 4) | ((ll >> 4) << 3) | (e & 7));
        wfrag[idx] = (_Float16)((k < KW_) ? Wl[nn * KW_ + k] : 0.f);
    }
    for (int idx = tid; idx < KT_ * 512; idx += 128) comb[idx] = (_Float16)0.f; // h0=0, K-pad=0
    for (int idx = tid; idx < 2 * 512;  idx += 128) hbuf[idx] = (_Float16)0.f;

    // per-lane gate biases (column hw*16+n of each gate, split order i,f,o,g)
    const float bi = bl[0 * HID_ + hw * 16 + n];
    const float bf = bl[1 * HID_ + hw * 16 + n];
    const float bo = bl[2 * HID_ + hw * 16 + n];
    const float bg = bl[3 * HID_ + hw * 16 + n];

    // classifier B fragments in registers: cols 0..9 = W_class, col 10 = W_dec
    v16h wc0, wc1; float bcl;
    {
        int half = l >> 4;
        #pragma unroll
        for (int e = 0; e < 16; ++e) {
            int kbase = ((e >> 3) << 4) | (half << 3) | (e & 7);
            int k0 = kbase, k1 = 32 + kbase;
            float v0 = (n < NCLS_) ? Wc[n * HID_ + k0] : ((n == 10) ? Wd[k0] : 0.f);
            float v1 = (n < NCLS_) ? Wc[n * HID_ + k1] : ((n == 10) ? Wd[k1] : 0.f);
            wc0[e] = (_Float16)v0;
            wc1[e] = (_Float16)v1;
        }
        bcl = (n < NCLS_) ? bc[n] : ((n == 10) ? bd[0] : 0.f);
    }

    // precomputed LDS write offsets for this lane's h column
    const int kc = 16 + hw * 16 + n;   // h column position in comb K-space
    const int comb_h = (kc >> 5) * 512 + eOfK(kc & 31) + 16 * (16 * hOfK(kc & 31) + mhi);
    const int kh = hw * 16 + n;        // h column position in classifier K-space
    const int hbuf_h = (kh >> 5) * 512 + eOfK(kh & 31) + 16 * (16 * hOfK(kh & 31) + mhi);

    float cst[8], pnd[8];
    #pragma unroll
    for (int r = 0; r < 8; ++r) { cst[r] = 0.f; pnd[r] = 1.f; }

    __syncthreads();

    #pragma unroll 1
    for (int t = 0; t < T_; ++t) {
        // ---- stage 1: stage x_t (16 rows x 16 ch) into comb K=0..15 ----
        #pragma unroll
        for (int j = 0; j < 2; ++j) {
            int ii = tid * 2 + j;
            int m = ii >> 4, k = ii & 15;
            const float* px = &x[((size_t)(b0 + m) * T_ + t) * CIN_ + k];
            float v = *px;
            if (k == 0 && t + 1 < T_) __builtin_prefetch(px + CIN_, 0, 0); // one per 64B row
            comb[eOfK(k) + 16 * (m + 16 * hOfK(k))] = (_Float16)v;
        }
        __syncthreads();

        // ---- stage 2: gates = [x_t | h] @ W_lstm^T ; one 16x16 tile per gate ----
        v16h a0 = *(const v16h*)(comb + 0 * 512 + l * 16);
        v16h a1 = *(const v16h*)(comb + 1 * 512 + l * 16);
        v16h a2 = *(const v16h*)(comb + 2 * 512 + l * 16);
        v8f acc[4];
        #pragma unroll
        for (int g = 0; g < 4; ++g) {
            int nt = g * 4 + hw;
            v8f c8 = {0.f, 0.f, 0.f, 0.f, 0.f, 0.f, 0.f, 0.f};
            v16h bw0 = *(const v16h*)(wfrag + (nt * 3 + 0) * 512 + l * 16);
            c8 = __builtin_amdgcn_wmma_f32_16x16x32_f16(false, a0, false, bw0, (short)0, c8, false, false);
            v16h bw1 = *(const v16h*)(wfrag + (nt * 3 + 1) * 512 + l * 16);
            c8 = __builtin_amdgcn_wmma_f32_16x16x32_f16(false, a1, false, bw1, (short)0, c8, false, false);
            v16h bw2 = *(const v16h*)(wfrag + (nt * 3 + 2) * 512 + l * 16);
            c8 = __builtin_amdgcn_wmma_f32_16x16x32_f16(false, a2, false, bw2, (short)0, c8, false, false);
            acc[g] = c8;
        }
        __syncthreads();   // all reads of comb complete before h overwrite

        // ---- stage 3: LSTM pointwise update, write h back (f16) ----
        #pragma unroll
        for (int r = 0; r < 8; ++r) {
            float iv = fast_sig (acc[0][r] + bi);
            float fv = fast_sig (acc[1][r] + bf);
            float ov = fast_sig (acc[2][r] + bo);
            float gv = fast_tanh(acc[3][r] + bg);
            float cv = fv * cst[r] + iv * gv;
            cst[r] = cv;
            float hv = ov * fast_tanh(cv);
            _Float16 hh = (_Float16)hv;
            comb[comb_h + r * 16] = hh;
            hbuf[hbuf_h + r * 16] = hh;
        }
        __syncthreads();   // h visible to classifier wave

        // ---- stage 4: classifier + decision head (wave 0 only) ----
        if (hw == 0) {
            v16h ha0 = *(const v16h*)(hbuf + 0 * 512 + l * 16);
            v16h ha1 = *(const v16h*)(hbuf + 1 * 512 + l * 16);
            v8f cc = {0.f, 0.f, 0.f, 0.f, 0.f, 0.f, 0.f, 0.f};
            cc = __builtin_amdgcn_wmma_f32_16x16x32_f16(false, ha0, false, wc0, (short)0, cc, false, false);
            cc = __builtin_amdgcn_wmma_f32_16x16x32_f16(false, ha1, false, wc1, (short)0, cc, false, false);
            #pragma unroll
            for (int r = 0; r < 8; ++r) {
                int m = r + mhi;
                float v = cc[r] + bcl;
                // log-softmax across classes (lanes n=0..9 of each 16-lane group)
                float mv = (n < NCLS_) ? v : -3.0e38f;
                #pragma unroll
                for (int s = 1; s < 16; s <<= 1) mv = fmaxf(mv, __shfl_xor(mv, s, 16));
                float ev = (n < NCLS_) ? __builtin_amdgcn_exp2f((v - mv) * LOG2E_) : 0.f;
                #pragma unroll
                for (int s = 1; s < 16; s <<= 1) ev += __shfl_xor(ev, s, 16);
                if (n < NCLS_) {
                    float lse = __builtin_amdgcn_logf(ev) * LN2_;   // ev >= 1, safe
                    out_logp[((size_t)(b0 + m) * T_ + t) * NCLS_ + n] = (v - mv) - lse;
                }
                if (n == 10) {  // decision head column; pnd recurrence lives in these lanes
                    float dec = fast_sig(v);
                    float pt  = (t == T_ - 1) ? pnd[r] : dec * pnd[r];
                    out_pt[(size_t)(b0 + m) * T_ + t] = pt;
                    pnd[r] *= (1.f - dec);
                }
            }
        }
    }
}

extern "C" void kernel_launch(void* const* d_in, const int* in_sizes, int n_in,
                              void* d_out, int out_size, void* d_ws, size_t ws_size,
                              hipStream_t stream) {
    const float* x  = (const float*)d_in[0];
    const float* Wl = (const float*)d_in[1];
    const float* bl = (const float*)d_in[2];
    const float* Wc = (const float*)d_in[3];
    const float* bc = (const float*)d_in[4];
    const float* Wd = (const float*)d_in[5];
    const float* bd = (const float*)d_in[6];
    float* out_logp = (float*)d_out;
    float* out_pt   = out_logp + (size_t)B_ * T_ * NCLS_;
    hipLaunchKernelGGL(convlstm_fused, dim3(B_ / 16), dim3(128), 0, stream,
                       x, Wl, bl, Wc, bc, Wd, bd, out_logp, out_pt);
}
